// StreamingFCBlock_62337155334726
// MI455X (gfx1250) — matrix operand order repaired and checked by reading, not compiled, not run on hardware
//
#include <hip/hip_runtime.h>
#include <hip/hip_bf16.h>
#include <stdint.h>

// ---------------------------------------------------------------------------
// Problem constants (match reference)
// ---------------------------------------------------------------------------
#define BATCH   32
#define TNEW    128
#define DMODEL  1024
#define NHEAD   16
#define DKH     64
#define WLCTX   70
#define KCONV   9
#define FFDIM   4096
#define D2      2048
#define MROWS   (BATCH*TNEW)        // 4096
#define TCONV   (TNEW + KCONV - 1)  // 136
#define MCONV   (BATCH*TCONV)       // 4352
#define KLALL   (WLCTX + TNEW)      // 198

typedef __attribute__((ext_vector_type(16))) __bf16 v16bf;
typedef __attribute__((ext_vector_type(8)))  float  v8f;

__device__ __forceinline__ float bf2f(unsigned short u) {
    union { unsigned u; float f; } c; c.u = ((unsigned)u) << 16; return c.f;
}
__device__ __forceinline__ unsigned short f2bf(float f) {
    union { float f; unsigned u; } c; c.f = f;
    unsigned r = c.u + 0x7fffu + ((c.u >> 16) & 1u);
    return (unsigned short)(r >> 16);
}
__device__ __forceinline__ float silu_f(float x) { return x / (1.f + __expf(-x)); }

// ---------------------------------------------------------------------------
// Async global->LDS staging (CDNA5 GLOBAL_LOAD_ASYNC_TO_LDS_B128, ASYNCcnt).
// The builtin's params are pointers to int4 (vector_size(16)) in AS1 / AS3.
// Guarded; falls back to a synchronous copy if the builtins are absent.
// ---------------------------------------------------------------------------
#if defined(__gfx1250__) && \
    __has_builtin(__builtin_amdgcn_global_load_async_to_lds_b128) && \
    __has_builtin(__builtin_amdgcn_s_wait_asynccnt)
#define USE_ASYNC_LDS 1
#else
#define USE_ASYNC_LDS 0
#endif

#if USE_ASYNC_LDS
typedef int vi4 __attribute__((vector_size(16)));
typedef __attribute__((address_space(1))) vi4 gvi4;
typedef __attribute__((address_space(3))) vi4 lvi4;
#endif

__device__ __forceinline__ void cp16_g2l(void* ldsdst, const void* gsrc) {
#if USE_ASYNC_LDS
    // Generic LDS address: low 32 bits are the LDS byte offset (ISA aperture rule).
    __builtin_amdgcn_global_load_async_to_lds_b128(
        (gvi4*)(uintptr_t)gsrc,
        (lvi4*)(unsigned)(uintptr_t)ldsdst, 0, 0);
#else
    *(uint4*)ldsdst = *(const uint4*)gsrc;
#endif
}
__device__ __forceinline__ void wait_stage() {
#if USE_ASYNC_LDS
    __builtin_amdgcn_s_wait_asynccnt(0);
#endif
}

// ---------------------------------------------------------------------------
// WMMA GEMM: C[M,N] = A[M,K](bf16) * Bt[N,K](bf16, pre-transposed) + bias.
// Block tile 128x128, 256 threads = 8 wave32 waves arranged 4(M) x 2(N);
// each wave computes 32x64 = 2x4 v_wmma_f32_16x16x32_bf16 tiles, K-step 32,
// double-buffered LDS fed by async b128 copies.
// ---------------------------------------------------------------------------
enum { EPI_BF16 = 0, EPI_SILU = 1, EPI_RES = 3, EPI_RESHALF = 4 };

template<int EPI>
__global__ __launch_bounds__(256) void gemm_bf16(
    const unsigned short* __restrict__ A,   // [M,K] row-major bf16
    const unsigned short* __restrict__ Bt,  // [N,K] row-major bf16 (B transposed)
    const float* __restrict__ bias,         // [N]
    const float* __restrict__ res,          // [M,N] f32 residual (EPI_RES*)
    float* __restrict__ outf,               // f32 out (EPI_RES*)
    unsigned short* __restrict__ outh,      // bf16 out (EPI_BF16/SILU)
    int M, int N, int K)
{
    __shared__ __align__(16) unsigned short lA[2][128 * 32];  // [m][k]
    __shared__ __align__(16) unsigned short lB[2][128 * 32];  // [n][k]

    const int tid = threadIdx.x;
    const int m0 = blockIdx.y * 128, n0 = blockIdx.x * 128;
    const int w  = tid >> 5, l = tid & 31;
    const int wm = w >> 1, wn = w & 1;       // 4 waves in M, 2 in N
    const int lr = l & 15, hi = l >> 4;

    union CF { v8f v; float f[8]; };
    CF acc[2][4];
#pragma unroll
    for (int mt = 0; mt < 2; ++mt)
#pragma unroll
        for (int nt = 0; nt < 4; ++nt)
#pragma unroll
            for (int r = 0; r < 8; ++r) acc[mt][nt].f[r] = 0.f;

    // Both tiles are 128 rows x 32 k of bf16: 512 x 16B chunks, 2 per thread.
    auto stage = [&](int buf, int kt) {
        const unsigned short* srcA = A  + (size_t)m0 * K + (size_t)kt * 32;
        const unsigned short* srcB = Bt + (size_t)n0 * K + (size_t)kt * 32;
#pragma unroll
        for (int i = 0; i < 2; ++i) {
            int c = tid + i * 256, r = c >> 2, cc = c & 3;
            cp16_g2l(&lA[buf][r * 32 + cc * 8], srcA + (size_t)r * K + cc * 8);
            cp16_g2l(&lB[buf][r * 32 + cc * 8], srcB + (size_t)r * K + cc * 8);
        }
    };

    stage(0, 0);
    wait_stage();
    __syncthreads();
    const int nk = K >> 5;
    for (int kt = 0; kt < nk; ++kt) {
        const int cur = kt & 1;
        if (kt + 1 < nk) stage(cur ^ 1, kt + 1);

        union AF { v16bf v; unsigned u[8]; } a[2], bm[4];
        // A fragment: 16-bit A 16x32 layout (lane-half selects K group)
#pragma unroll
        for (int mt = 0; mt < 2; ++mt) {
            int row = wm * 32 + mt * 16 + lr;
#pragma unroll
            for (int j = 0; j < 8; ++j) {
                int kp = hi * 4 + j + ((j >= 4) ? 4 : 0);  // K pair index
                a[mt].u[j] = *(const unsigned*)&lA[cur][row * 32 + kp * 2];
            }
        }
        // B fragment: 32x16 B; same K order as A (lanes 0-15 K=0..7,16..23; lanes 16-31 K=8..15,24..31).
        // lB is n-major so each fragment is two 16B-contiguous groups.
#pragma unroll
        for (int nt = 0; nt < 4; ++nt) {
            int n  = wn * 64 + nt * 16 + lr;
            int kb = hi * 16;
#pragma unroll
            for (int j = 0; j < 8; ++j)
                bm[nt].u[j] = *(const unsigned*)&lB[cur][n * 32 + (kb >> 1) + (j + ((j >= 4) ? 4 : 0)) * 2];
        }
#pragma unroll
        for (int mt = 0; mt < 2; ++mt)
#pragma unroll
            for (int nt = 0; nt < 4; ++nt)
                acc[mt][nt].v = __builtin_amdgcn_wmma_f32_16x16x32_bf16(
                    false, a[mt].v, false, bm[nt].v,
                    (short)0, acc[mt][nt].v, false, false);
        wait_stage();
        __syncthreads();
    }

    // Epilogue. C/D layout: VGPR r -> row = r + 8*hi; col = lr.
#pragma unroll
    for (int nt = 0; nt < 4; ++nt) {
        int col = n0 + wn * 64 + nt * 16 + lr;
        float bv = bias[col];
#pragma unroll
        for (int mt = 0; mt < 2; ++mt) {
#pragma unroll
            for (int r = 0; r < 8; ++r) {
                int row = m0 + wm * 32 + mt * 16 + r + hi * 8;
                size_t idx = (size_t)row * N + col;
                float v = acc[mt][nt].f[r] + bv;
                if (EPI == EPI_BF16)        outh[idx] = f2bf(v);
                else if (EPI == EPI_SILU)   outh[idx] = f2bf(silu_f(v));
                else if (EPI == EPI_RES)    outf[idx] = res[idx] + v;
                else /* EPI_RESHALF */      outf[idx] = res[idx] + 0.5f * v;
            }
        }
    }
}

// ---------------------------------------------------------------------------
// LayerNorm over D=1024, one block per row, bf16 output.
// pad remaps output rows: orow = row + pad*(row/128 + 1) (ring-buffer concat).
// ---------------------------------------------------------------------------
__global__ __launch_bounds__(256) void ln_kernel(
    const float* __restrict__ x, const float* __restrict__ g,
    const float* __restrict__ bb, unsigned short* __restrict__ out, int pad)
{
    const int row = blockIdx.x, tid = threadIdx.x;
    const float* src = x + (size_t)row * DMODEL;
    float4 v = *(const float4*)(src + tid * 4);
    float s  = v.x + v.y + v.z + v.w;
    float s2 = v.x * v.x + v.y * v.y + v.z * v.z + v.w * v.w;
    __shared__ float r1[256], r2[256];
    r1[tid] = s; r2[tid] = s2;
    __syncthreads();
    for (int off = 128; off > 0; off >>= 1) {
        if (tid < off) { r1[tid] += r1[tid + off]; r2[tid] += r2[tid + off]; }
        __syncthreads();
    }
    float mean = r1[0] * (1.f / DMODEL);
    float var  = r2[0] * (1.f / DMODEL) - mean * mean;
    float rstd = rsqrtf(var + 1e-5f);
    int orow = row + pad * ((row >> 7) + 1);
    unsigned short* o = out + (size_t)orow * DMODEL + tid * 4;
    float4 gv = *(const float4*)(g + tid * 4);
    float4 bv = *(const float4*)(bb + tid * 4);
    o[0] = f2bf((v.x - mean) * rstd * gv.x + bv.x);
    o[1] = f2bf((v.y - mean) * rstd * gv.y + bv.y);
    o[2] = f2bf((v.z - mean) * rstd * gv.z + bv.z);
    o[3] = f2bf((v.w - mean) * rstd * gv.w + bv.w);
}

// ---------------------------------------------------------------------------
// Attention: one block per (b,h); K/V (ctx fp32 + new bf16) staged in LDS,
// 128 threads = one query row each, single-pass online softmax.
// ---------------------------------------------------------------------------
__global__ __launch_bounds__(128) void attn_kernel(
    const float* __restrict__ Kctx, const float* __restrict__ Vctx,
    const unsigned short* __restrict__ Qb, const unsigned short* __restrict__ Kb,
    const unsigned short* __restrict__ Vb, unsigned short* __restrict__ Ob)
{
    __shared__ unsigned short Kl[KLALL * DKH];
    __shared__ unsigned short Vl[KLALL * DKH];
    const int bh = blockIdx.x, b = bh >> 4, h = bh & 15;
    const int tid = threadIdx.x;
    const float* kc = Kctx + (size_t)bh * WLCTX * DKH;
    const float* vc = Vctx + (size_t)bh * WLCTX * DKH;
    for (int i = tid; i < KLALL * DKH; i += 128) {
        int r = i >> 6, d = i & 63;
        unsigned short kv, vv;
        if (r < WLCTX) { kv = f2bf(kc[r * DKH + d]); vv = f2bf(vc[r * DKH + d]); }
        else {
            size_t idx = ((size_t)(b * TNEW + (r - WLCTX))) * DMODEL + h * DKH + d;
            kv = Kb[idx]; vv = Vb[idx];
        }
        Kl[i] = kv; Vl[i] = vv;
    }
    __syncthreads();

    const int t = tid;
    size_t qbase = ((size_t)(b * TNEW + t)) * DMODEL + h * DKH;
    float q[DKH], accv[DKH];
#pragma unroll
    for (int d = 0; d < DKH; ++d) { q[d] = bf2f(Qb[qbase + d]); accv[d] = 0.f; }
    float m = -1e30f, ssum = 0.f;
    for (int k = 0; k < KLALL; ++k) {
        float s = 0.f;
#pragma unroll
        for (int d = 0; d < DKH; ++d) s += q[d] * bf2f(Kl[k * DKH + d]);
        s *= 0.125f;                              // 1/sqrt(64)
        float mn = fmaxf(m, s);
        float corr = __expf(m - mn);
        float p = __expf(s - mn);
        ssum = ssum * corr + p;
#pragma unroll
        for (int d = 0; d < DKH; ++d) accv[d] = accv[d] * corr + p * bf2f(Vl[k * DKH + d]);
        m = mn;
    }
    float inv = 1.f / ssum;
#pragma unroll
    for (int d = 0; d < DKH; ++d) Ob[qbase + d] = f2bf(accv[d] * inv);
}

// ---------------------------------------------------------------------------
// Depthwise conv (k=9, SAME on length-136 seq, keep last 128) + BN + SiLU.
// y1: bf16 [B, 136, 2D]; z: bf16 [B*T, 2D]
// ---------------------------------------------------------------------------
__global__ __launch_bounds__(256) void dwconv_kernel(
    const unsigned short* __restrict__ y1, const float* __restrict__ dw_w,
    const float* __restrict__ dw_b, const float* __restrict__ bn_g,
    const float* __restrict__ bn_b, const float* __restrict__ bn_mean,
    const float* __restrict__ bn_var, unsigned short* __restrict__ z)
{
    size_t i = (size_t)blockIdx.x * 256 + threadIdx.x;  // over MROWS*D2
    int e = (int)(i % D2);
    int t = (int)((i / D2) % TNEW);
    int b = (int)(i / ((size_t)D2 * TNEW));
    int p = t + (KCONV - 1);                            // output pos in [8,136)
    float acc = 0.f;
#pragma unroll
    for (int kk = 0; kk < KCONV; ++kk) {
        int r = p - 4 + kk;                             // 4..139; >=136 is zero-pad
        if (r < TCONV)
            acc += bf2f(y1[((size_t)(b * TCONV + r)) * D2 + e]) * dw_w[e * KCONV + kk];
    }
    acc += dw_b[e];
    acc = (acc - bn_mean[e]) * rsqrtf(bn_var[e] + 1e-5f) * bn_g[e] + bn_b[e];
    z[i] = f2bf(silu_f(acc));
}

// ---------------------------------------------------------------------------
// Conversions
// ---------------------------------------------------------------------------
__global__ __launch_bounds__(256) void cvt_weight(
    const float* __restrict__ s, unsigned short* __restrict__ d,
    int R, int C, int T)  // T: write transposed [C,R]
{
    size_t i = (size_t)blockIdx.x * 256 + threadIdx.x;
    if (i >= (size_t)R * C) return;
    int r = (int)(i / C), c = (int)(i % C);
    unsigned short v = f2bf(s[i]);
    if (T) d[(size_t)c * R + r] = v; else d[i] = v;
}

__global__ __launch_bounds__(256) void cvt_plain(
    const float* __restrict__ s, unsigned short* __restrict__ d, size_t n)
{
    size_t i = (size_t)blockIdx.x * 256 + threadIdx.x;
    if (i < n) d[i] = f2bf(s[i]);
}

// conv_buf [B,8,D] -> bf16 rows 0..7 of each batch's 136-row y_in block
__global__ __launch_bounds__(256) void cvt_convbuf(
    const float* __restrict__ s, unsigned short* __restrict__ d)
{
    size_t i = (size_t)blockIdx.x * 256 + threadIdx.x;
    if (i >= (size_t)BATCH * 8 * DMODEL) return;
    size_t b = i >> 13, rem = i & 8191;     // 8*1024 = 8192
    d[b * (size_t)TCONV * DMODEL + rem] = f2bf(s[i]);
}

// ---------------------------------------------------------------------------
// Host-side orchestration. Workspace layout (requires ~114 MB):
//   [0,8M)     fc1^T bf16 [4096,1024]   [8M,16M)  fc2^T bf16 [1024,4096]
//   [16M..24M) q/k/v/o ^T bf16 [1024,1024] each (2MB apiece)
//   [24M,28M)  pw1 bf16 [2048,1024]     [28M,32M) pw2 bf16 [1024,2048]
//   (all weights stored [N,K] so GEMM B-tiles are contiguous b128 chunks)
//   [32M,48M)  x f32 [4096,1024]
//   [48M,64M)  bufA bf16 (LN outs / y_in / x_bf16)
//   [64M,96M)  bufB bf16 (h1|h2 / Q+K+V+attn / z)
//   [96M,114M) bufC bf16 (pw1 out y1 [4352,2048])
// ---------------------------------------------------------------------------
extern "C" void kernel_launch(void* const* d_in, const int* in_sizes, int n_in,
                              void* d_out, int out_size, void* d_ws, size_t ws_size,
                              hipStream_t stream)
{
    const float* x_new    = (const float*)d_in[0];
    const float* K_ctx    = (const float*)d_in[1];
    const float* V_ctx    = (const float*)d_in[2];
    const float* conv_buf = (const float*)d_in[3];
    const float* ln_ff1_g = (const float*)d_in[4];
    const float* ln_ff1_b = (const float*)d_in[5];
    const float* fc1_w    = (const float*)d_in[6];
    const float* fc1_b    = (const float*)d_in[7];
    const float* fc2_w    = (const float*)d_in[8];
    const float* fc2_b    = (const float*)d_in[9];
    const float* ln_at_g  = (const float*)d_in[10];
    const float* ln_at_b  = (const float*)d_in[11];
    const float* q_w      = (const float*)d_in[12];
    const float* q_b      = (const float*)d_in[13];
    const float* k_w      = (const float*)d_in[14];
    const float* k_b      = (const float*)d_in[15];
    const float* v_w      = (const float*)d_in[16];
    const float* v_b      = (const float*)d_in[17];
    const float* o_w      = (const float*)d_in[18];
    const float* o_b      = (const float*)d_in[19];
    const float* ln_cv_g  = (const float*)d_in[20];
    const float* ln_cv_b  = (const float*)d_in[21];
    const float* pw1_w    = (const float*)d_in[22];
    const float* pw1_b    = (const float*)d_in[23];
    const float* dw_w     = (const float*)d_in[24];
    const float* dw_b     = (const float*)d_in[25];
    const float* bn_g     = (const float*)d_in[26];
    const float* bn_b     = (const float*)d_in[27];
    const float* bn_mean  = (const float*)d_in[28];
    const float* bn_var   = (const float*)d_in[29];
    const float* pw2_w    = (const float*)d_in[30];
    const float* pw2_b    = (const float*)d_in[31];
    float* out = (float*)d_out;

    constexpr size_t MB = 1024ull * 1024ull;
    char* ws = (char*)d_ws;
    unsigned short* wfc1 = (unsigned short*)(ws + 0 * MB);
    unsigned short* wfc2 = (unsigned short*)(ws + 8 * MB);
    unsigned short* wq   = (unsigned short*)(ws + 16 * MB);
    unsigned short* wk   = (unsigned short*)(ws + 18 * MB);
    unsigned short* wv   = (unsigned short*)(ws + 20 * MB);
    unsigned short* wo   = (unsigned short*)(ws + 22 * MB);
    unsigned short* wpw1 = (unsigned short*)(ws + 24 * MB);
    unsigned short* wpw2 = (unsigned short*)(ws + 28 * MB);
    float*          xst  = (float*)         (ws + 32 * MB);
    unsigned short* bufA = (unsigned short*)(ws + 48 * MB);
    unsigned short* bufB = (unsigned short*)(ws + 64 * MB);
    unsigned short* bufC = (unsigned short*)(ws + 96 * MB);
    unsigned short* Qb   = bufB;
    unsigned short* Kb   = (unsigned short*)(ws + 72 * MB);
    unsigned short* Vb   = (unsigned short*)(ws + 80 * MB);
    unsigned short* Ab   = (unsigned short*)(ws + 88 * MB);  // attn out

    const dim3 blk(256);

    // --- weight conversion to [N,K] bf16 (each call; deterministic) ---
    cvt_weight<<<16384, blk, 0, stream>>>(fc1_w, wfc1, 1024, 4096, 1);
    cvt_weight<<<16384, blk, 0, stream>>>(fc2_w, wfc2, 4096, 1024, 1);
    cvt_weight<<<4096,  blk, 0, stream>>>(q_w,   wq,   1024, 1024, 1);
    cvt_weight<<<4096,  blk, 0, stream>>>(k_w,   wk,   1024, 1024, 1);
    cvt_weight<<<4096,  blk, 0, stream>>>(v_w,   wv,   1024, 1024, 1);
    cvt_weight<<<4096,  blk, 0, stream>>>(o_w,   wo,   1024, 1024, 1);
    cvt_weight<<<8192,  blk, 0, stream>>>(pw1_w, wpw1, 2048, 1024, 0);  // already [N,K]
    cvt_weight<<<8192,  blk, 0, stream>>>(pw2_w, wpw2, 1024, 2048, 0);  // already [N,K]

    // --- FF/2 #1: x = x_new + 0.5*ff(ln(x_new)) ---
    ln_kernel<<<MROWS, blk, 0, stream>>>(x_new, ln_ff1_g, ln_ff1_b, bufA, 0);
    gemm_bf16<EPI_SILU><<<dim3(FFDIM / 128, MROWS / 128), blk, 0, stream>>>(
        bufA, wfc1, fc1_b, nullptr, nullptr, bufB, MROWS, FFDIM, DMODEL);
    gemm_bf16<EPI_RESHALF><<<dim3(DMODEL / 128, MROWS / 128), blk, 0, stream>>>(
        bufB, wfc2, fc2_b, x_new, xst, nullptr, MROWS, DMODEL, FFDIM);

    // --- attention ---
    ln_kernel<<<MROWS, blk, 0, stream>>>(xst, ln_at_g, ln_at_b, bufA, 0);
    gemm_bf16<EPI_BF16><<<dim3(DMODEL / 128, MROWS / 128), blk, 0, stream>>>(
        bufA, wq, q_b, nullptr, nullptr, Qb, MROWS, DMODEL, DMODEL);
    gemm_bf16<EPI_BF16><<<dim3(DMODEL / 128, MROWS / 128), blk, 0, stream>>>(
        bufA, wk, k_b, nullptr, nullptr, Kb, MROWS, DMODEL, DMODEL);
    gemm_bf16<EPI_BF16><<<dim3(DMODEL / 128, MROWS / 128), blk, 0, stream>>>(
        bufA, wv, v_b, nullptr, nullptr, Vb, MROWS, DMODEL, DMODEL);
    attn_kernel<<<BATCH * NHEAD, dim3(128), 0, stream>>>(K_ctx, V_ctx, Qb, Kb, Vb, Ab);
    gemm_bf16<EPI_RES><<<dim3(DMODEL / 128, MROWS / 128), blk, 0, stream>>>(
        Ab, wo, o_b, xst, xst, nullptr, MROWS, DMODEL, DMODEL);

    // --- conv module ---
    cvt_convbuf<<<1024, blk, 0, stream>>>(conv_buf, bufA);                // rows 0..7/batch
    ln_kernel<<<MROWS, blk, 0, stream>>>(xst, ln_cv_g, ln_cv_b, bufA, 8); // rows 8..135
    gemm_bf16<EPI_BF16><<<dim3(D2 / 128, MCONV / 128), blk, 0, stream>>>(
        bufA, wpw1, pw1_b, nullptr, nullptr, bufC, MCONV, D2, DMODEL);
    dwconv_kernel<<<(MROWS * D2) / 256, blk, 0, stream>>>(
        bufC, dw_w, dw_b, bn_g, bn_b, bn_mean, bn_var, bufB);             // z in bufB
    gemm_bf16<EPI_RES><<<dim3(DMODEL / 128, MROWS / 128), blk, 0, stream>>>(
        bufB, wpw2, pw2_b, xst, xst, nullptr, MROWS, DMODEL, D2);

    // --- FF/2 #2 (no LN): out = x + 0.5*ff(x) ---
    cvt_plain<<<16384, blk, 0, stream>>>(xst, bufA, (size_t)MROWS * DMODEL);
    gemm_bf16<EPI_SILU><<<dim3(FFDIM / 128, MROWS / 128), blk, 0, stream>>>(
        bufA, wfc1, fc1_b, nullptr, nullptr, bufB, MROWS, FFDIM, DMODEL);
    gemm_bf16<EPI_RESHALF><<<dim3(DMODEL / 128, MROWS / 128), blk, 0, stream>>>(
        bufB, wfc2, fc2_b, xst, out, nullptr, MROWS, DMODEL, FFDIM);
}
